// WindowAttention_23210003268206
// MI455X (gfx1250) — matrix operand order repaired and checked by reading, not compiled, not run on hardware
//
#include <hip/hip_runtime.h>
#include <hip/hip_bf16.h>

typedef __attribute__((ext_vector_type(16))) _Float16 v16h;
typedef __attribute__((ext_vector_type(8)))  float    v8f;

#define NUM_HEADS 8
#define M_WIN     64
#define D_HEAD    64
#define TBL_ROWS  16129   // (2*64-1)^2
#define LDR       72      // padded LDS row stride (halves) -> conflict-free

__global__ __launch_bounds__(128)
void winattn_wmma_kernel(const float* __restrict__ q,
                         const float* __restrict__ k,
                         const float* __restrict__ v,
                         const float* __restrict__ table,
                         float* __restrict__ out)
{
    __shared__ _Float16 ldsK[M_WIN * LDR];   // K  row-major [n][d]  (f16)
    __shared__ _Float16 ldsVT[D_HEAD * LDR]; // V transposed [d][n]  (f16)
    __shared__ _Float16 ldsP[M_WIN * LDR];   // softmax probs [m][n] (f16)

    const int blk  = blockIdx.x;        // b*NUM_HEADS + h
    const int h    = blk & (NUM_HEADS - 1);
    const int bwin = blk >> 3;
    const long base = (long)blk * (M_WIN * D_HEAD);

    const int tid  = threadIdx.x;       // 0..127
    const int wave = tid >> 5;          // 0..3  : 16-row query strip
    const int lane = tid & 31;
    const int l15  = lane & 15;
    const int sub  = lane >> 4;         // which half of the wave (K-range select)

    // ---- cooperative K/V staging: global f32 -> LDS f16 -------------------
    for (int i = tid; i < M_WIN * D_HEAD; i += 128) {
        const int r = i >> 6;           // token index n
        const int c = i & 63;           // feature index d
        ldsK [r * LDR + c] = (_Float16)k[base + i];
        ldsVT[c * LDR + r] = (_Float16)v[base + i];  // transposed for O-GEMM B
    }
    __syncthreads();

    // ---- Q A-fragments (scale folded in; 0.125 exact in f16) --------------
    const float scale = 0.125f;  // 64^-0.5
    const int   mrowA = wave * 16 + l15;          // A-layout: row = lane&15
    const float* qrow = q + base + (long)mrowA * D_HEAD;
    v16h aq[2];
    #pragma unroll
    for (int s = 0; s < 2; ++s) {
        #pragma unroll
        for (int j = 0; j < 16; ++j) {
            const int kk = s * 32 + sub * 16 + j;
            aq[s][j] = (_Float16)(qrow[kk] * scale);
        }
    }

    // ---- S = Q K^T (+bias): 4 column tiles, K=64 split as 2x32 ------------
    v8f c[4];
    #pragma unroll
    for (int t = 0; t < 4; ++t) {
        v8f acc = {};
        #pragma unroll
        for (int s = 0; s < 2; ++s) {
            const _Float16* kp = &ldsK[(t * 16 + l15) * LDR + s * 32 + sub * 16];
            v16h b;
            #pragma unroll
            for (int j = 0; j < 16; ++j) b[j] = kp[j];
            acc = __builtin_amdgcn_wmma_f32_16x16x32_f16(
                      false, aq[s], false, b, (short)0, acc, false, false);
        }
        // relative-position bias, torch negative-index wrap via mod
        #pragma unroll
        for (int i = 0; i < 8; ++i) {
            const int mr = wave * 16 + i + 8 * sub;   // query row (C-layout)
            const int nc = t * 16 + l15;              // key col
            int idx = 128 * mr + nc - 4064;           // (mr-32)*128+(nc-32)+64
            if (idx < 0) idx += TBL_ROWS;
            acc[i] += table[idx * NUM_HEADS + h];
        }
        c[t] = acc;
    }

    // ---- row softmax: rows live inside a 16-lane half-group ---------------
    float rmax[8], rsum[8];
    #pragma unroll
    for (int i = 0; i < 8; ++i) {
        float mx = fmaxf(fmaxf(c[0][i], c[1][i]), fmaxf(c[2][i], c[3][i]));
        #pragma unroll
        for (int d = 1; d < 16; d <<= 1) mx = fmaxf(mx, __shfl_xor(mx, d, 32));
        rmax[i] = mx;
    }
    #pragma unroll
    for (int i = 0; i < 8; ++i) {
        float sum = 0.f;
        #pragma unroll
        for (int t = 0; t < 4; ++t) {
            const float e = __expf(c[t][i] - rmax[i]);
            c[t][i] = e;
            sum += e;
        }
        #pragma unroll
        for (int d = 1; d < 16; d <<= 1) sum += __shfl_xor(sum, d, 32);
        rsum[i] = sum;
    }

    // ---- P -> LDS (f16), wave-private strip: no cross-wave barrier --------
    #pragma unroll
    for (int t = 0; t < 4; ++t) {
        #pragma unroll
        for (int i = 0; i < 8; ++i) {
            const int mr = wave * 16 + i + 8 * sub;
            const int nc = t * 16 + l15;
            ldsP[mr * LDR + nc] = (_Float16)(c[t][i] / rsum[i]);
        }
    }

    // ---- O = P V: A from LDS P strip, B from transposed V -----------------
    v16h ap[2];
    const _Float16* prow = &ldsP[(wave * 16 + l15) * LDR];
    #pragma unroll
    for (int s = 0; s < 2; ++s) {
        #pragma unroll
        for (int j = 0; j < 16; ++j) {
            const int kk = s * 32 + sub * 16 + j;
            ap[s][j] = prow[kk];
        }
    }

    #pragma unroll
    for (int t = 0; t < 4; ++t) {
        v8f acc = {};
        #pragma unroll
        for (int s = 0; s < 2; ++s) {
            const _Float16* vp = &ldsVT[(t * 16 + l15) * LDR + s * 32 + sub * 16];
            v16h b;
            #pragma unroll
            for (int j = 0; j < 16; ++j) b[j] = vp[j];
            acc = __builtin_amdgcn_wmma_f32_16x16x32_f16(
                      false, ap[s], false, b, (short)0, acc, false, false);
        }
        // store with fused transpose(0,2,1,3)+reshape+roll:
        //   w-axis roll by -32 on size-8 axis is identity;
        //   channel roll: ch = (h*64 + d - 32) mod 512
        #pragma unroll
        for (int i = 0; i < 8; ++i) {
            const int mr = wave * 16 + i + 8 * sub;   // token m in window
            const int dd = t * 16 + l15;              // feature d
            const int ch = (h * D_HEAD + dd - 32) & 511;
            out[(long)bwin * 32768 + mr * 512 + ch] = acc[i];
        }
    }
}

extern "C" void kernel_launch(void* const* d_in, const int* in_sizes, int n_in,
                              void* d_out, int out_size, void* d_ws, size_t ws_size,
                              hipStream_t stream) {
    const float* q     = (const float*)d_in[0];
    const float* k     = (const float*)d_in[1];
    const float* v     = (const float*)d_in[2];
    const float* table = (const float*)d_in[3];
    // d_in[4] = h (8), d_in[5] = w (8): fixed by problem geometry above.

    const int BH = in_sizes[0] / (M_WIN * D_HEAD);   // B_WIN * NUM_HEADS = 16384
    winattn_wmma_kernel<<<BH, 128, 0, stream>>>(q, k, v, table, (float*)d_out);
}
